// DecisionTransformer_23897198035165
// MI455X (gfx1250) — matrix operand order repaired and checked by reading, not compiled, not run on hardware
//
#include <hip/hip_runtime.h>
#include <hip/hip_bf16.h>

// ---------------------------------------------------------------------------
// Decision Transformer forward for gfx1250 (MI455X).
// wave32 + WMMA f16 for all GEMMs; Tensor Data Mover (TDM) streams GEMM tiles
// into LDS (double-buffered) with DMA-side padding producing the bank-
// conflict-free 80B row pitch the WMMA fragment loads expect.
// ---------------------------------------------------------------------------

typedef _Float16 h16;
typedef __attribute__((ext_vector_type(8)))  _Float16 h8;
typedef __attribute__((ext_vector_type(16))) _Float16 v16h;
typedef __attribute__((ext_vector_type(8)))  float    v8f;
typedef unsigned int u32x4 __attribute__((ext_vector_type(4)));
typedef int          i32x4 __attribute__((ext_vector_type(4)));
typedef int          i32x8 __attribute__((ext_vector_type(8)));

#define C_DIM   1280
#define NHEAD   20
#define DHEAD   64
#define NLAYER  6
#define PTOK    36
#define TPS     39
#define TT      4
#define BB      32
#define SEQ     156              // TPS * TT
#define MROWS   (BB * SEQ)       // 4992  (multiple of 64)
#define PROWS   (BB * TT * PTOK) // 4608  (multiple of 64)
#define KCONV   800              // 784 padded to multiple of 32
#define FF      5120

#if __has_builtin(__builtin_amdgcn_tensor_load_to_lds) && \
    __has_builtin(__builtin_amdgcn_s_wait_tensorcnt)
#define HAVE_TDM 1
#else
#define HAVE_TDM 0
#endif

// Build a 16-wide f16 vector from two aligned 8-wide halves.
__device__ __forceinline__ v16h mk16(h8 lo, h8 hi) {
    union { v16h v; h8 h[2]; } u;
    u.h[0] = lo; u.h[1] = hi;
    return u.v;
}

#if HAVE_TDM
// Issue one TDM 2D tile load: f16 tile (rows x 32 elems) from a row-major
// [rows_total x Ktot] f16 tensor into LDS, DMA-padding each 64B row with 16B
// so LDS row pitch is 80B (40 halves).
__device__ __forceinline__ void tdm_load_tile_f16(const h16* gsrc, void* lds,
                                                  unsigned rows, unsigned Ktot) {
    unsigned long long ga = (unsigned long long)(uintptr_t)gsrc; // 57-bit global
    unsigned lo = (unsigned)(uintptr_t)lds;                      // LDS byte offset
    u32x4 g0;
    g0[0] = 1u;                                        // count=1, user descriptor
    g0[1] = lo;                                        // lds_addr
    g0[2] = (unsigned)ga;                              // global_addr[31:0]
    g0[3] = ((unsigned)(ga >> 32) & 0x01FFFFFFu)       // global_addr[56:32]
            | 0x80000000u;                             // type=2 ("image")
    i32x8 g1;
    // data_size=2B (code 1), pad_enable, pad_interval=16 DWORDs (code 3),
    // pad_amount=4 DWORDs (code 3), workgroup_mask=0
    g1[0] = (int)((1u << 16) | (1u << 20) | (3u << 22) | (3u << 25));
    g1[1] = (int)((Ktot & 0xFFFFu) << 16);             // tensor_dim0[15:0]
    g1[2] = (int)(((Ktot >> 16) & 0xFFFFu) | ((rows & 0xFFFFu) << 16)); // td0 hi, td1 lo
    g1[3] = (int)(((rows >> 16) & 0xFFFFu) | (32u << 16)); // td1 hi, tile_dim0=32
    g1[4] = (int)(rows & 0xFFFFu);                     // tile_dim1=rows, tile_dim2=0
    g1[5] = (int)Ktot;                                 // tensor_dim0_stride[31:0]
    g1[6] = 0;                                         // stride0 hi, stride1 lo
    g1[7] = 0;                                         // stride1 hi
    i32x4 z4 = {0, 0, 0, 0};
#if defined(__clang_major__) && (__clang_major__ >= 23)
    i32x8 z8 = {0, 0, 0, 0, 0, 0, 0, 0};
    __builtin_amdgcn_tensor_load_to_lds(g0, g1, z4, z4, z8, 0);
#else
    __builtin_amdgcn_tensor_load_to_lds(g0, g1, z4, z4, 0);
#endif
}
#endif // HAVE_TDM

// Attention mask: causal OR same-timestep obs block, with key mask
// (ret tokens of t>0 are masked out); k >= SEQ is padding -> masked.
__device__ __forceinline__ bool att_allowed(int q, int k) {
    if (k >= SEQ) return false;
    int kt = k / TPS, kr = k % TPS;
    if (kr == PTOK && kt > 0) return false;       // masked ret token key
    if (k <= q) return true;                      // causal
    int qt = q / TPS, qr = q % TPS;
    return (qt == kt) && (qr < PTOK) && (kr < PTOK); // obs block full-attn
}

// ---------------------------------------------------------------------------
// Weight conversion: f32 [K][N] -> f16 [N][K] (transposed, row = output col)
// ---------------------------------------------------------------------------
__global__ void k_transpose(const float* __restrict__ src, h16* __restrict__ dst,
                            int K, int N) {
    int n = blockIdx.x;
    for (int k = threadIdx.x; k < K; k += blockDim.x)
        dst[(size_t)n * K + k] = (h16)src[(size_t)k * N + n];
}

// conv_w (1280, 4*14*14=784) is already output-major: just pad K to 800.
__global__ void k_convw(const float* __restrict__ src, h16* __restrict__ dst) {
    int o = blockIdx.x;
    for (int k = threadIdx.x; k < KCONV; k += blockDim.x)
        dst[(size_t)o * KCONV + k] = (h16)((k < 784) ? src[(size_t)o * 784 + k] : 0.0f);
}

// ---------------------------------------------------------------------------
// Patch matrix: obs int32 (B,T,4,84,84)/255 -> P f16 [4608][800]
// ---------------------------------------------------------------------------
__global__ void k_patchify(const int* __restrict__ obs, h16* __restrict__ P) {
    int r = blockIdx.x;
    int b = r / (TT * PTOK);
    int rem = r % (TT * PTOK);
    int t = rem / PTOK, p = rem % PTOK;
    int py = p / 6, px = p % 6;
    const int* ob = obs + (size_t)(b * TT + t) * 4 * 84 * 84;
    for (int col = threadIdx.x; col < KCONV; col += blockDim.x) {
        float v = 0.0f;
        if (col < 784) {
            int c = col / 196, rc = col % 196, ky = rc / 14, kx = rc % 14;
            v = (float)ob[(size_t)c * 7056 + (py * 14 + ky) * 84 + (px * 14 + kx)] * (1.0f / 255.0f);
        }
        P[(size_t)r * KCONV + col] = (h16)v;
    }
}

// ---------------------------------------------------------------------------
// Token embedding rows of x (ret/act/rew); ret token masked to 0 for t>0.
// ---------------------------------------------------------------------------
__global__ void k_embed(const int* __restrict__ rtg, const int* __restrict__ act,
                        const int* __restrict__ rew,
                        const float* __restrict__ ret_tab,
                        const float* __restrict__ act_tab,
                        const float* __restrict__ rew_tab,
                        float* __restrict__ x) {
    int bt = blockIdx.x;
    int b = bt / TT, t = bt % TT;
    int ri = rtg[bt]; ri = ri < -20 ? -20 : (ri > 100 ? 100 : ri); ri += 20;
    int ai = act[bt];
    int rv = rew[bt];
    int si = (rv > 0) - (rv < 0) + 1;
    size_t base = ((size_t)b * SEQ + (size_t)t * TPS) * C_DIM;
    for (int c = threadIdx.x; c < C_DIM; c += blockDim.x) {
        x[base + (size_t)(PTOK + 0) * C_DIM + c] = (t == 0) ? ret_tab[(size_t)ri * C_DIM + c] : 0.0f;
        x[base + (size_t)(PTOK + 1) * C_DIM + c] = act_tab[(size_t)ai * C_DIM + c];
        x[base + (size_t)(PTOK + 2) * C_DIM + c] = rew_tab[(size_t)si * C_DIM + c];
    }
}

// ---------------------------------------------------------------------------
// LayerNorm: x f32 row -> f16 normalized row (GEMM A input)
// ---------------------------------------------------------------------------
__global__ __launch_bounds__(256)
void k_ln(const float* __restrict__ x, const float* __restrict__ g,
          const float* __restrict__ bta, h16* __restrict__ out) {
    __shared__ float red[256];
    int row = blockIdx.x;
    const float* xr = x + (size_t)row * C_DIM;
    float s = 0.0f, s2 = 0.0f;
    for (int c = threadIdx.x; c < C_DIM; c += 256) {
        float v = xr[c]; s += v; s2 += v * v;
    }
    red[threadIdx.x] = s; __syncthreads();
    for (int st = 128; st > 0; st >>= 1) {
        if (threadIdx.x < st) red[threadIdx.x] += red[threadIdx.x + st];
        __syncthreads();
    }
    float mean = red[0] * (1.0f / C_DIM);
    __syncthreads();
    red[threadIdx.x] = s2; __syncthreads();
    for (int st = 128; st > 0; st >>= 1) {
        if (threadIdx.x < st) red[threadIdx.x] += red[threadIdx.x + st];
        __syncthreads();
    }
    float var = red[0] * (1.0f / C_DIM) - mean * mean;
    float rstd = rsqrtf(var + 1e-5f);
    for (int c = threadIdx.x; c < C_DIM; c += 256)
        out[(size_t)row * C_DIM + c] = (h16)((xr[c] - mean) * rstd * g[c] + bta[c]);
}

// ---------------------------------------------------------------------------
// One 64x128 x K=32 WMMA step from staged LDS tiles (pitch 40 halves).
// ---------------------------------------------------------------------------
__device__ __forceinline__ void gemm_step(const h16* __restrict__ As,
                                          const h16* __restrict__ Bs,
                                          int wm, int wn, int mfr,
                                          int klo, int kStart, v8f acc[2][2]) {
    v16h af[2], bf[2];
    for (int f = 0; f < 2; ++f) {
        int mr = wm * 32 + f * 16 + mfr;
        af[f] = mk16(*(const h8*)&As[mr * 40 + kStart],
                     *(const h8*)&As[mr * 40 + kStart + 8]);
        int nr = wn * 32 + f * 16 + mfr;
        bf[f] = mk16(*(const h8*)&Bs[nr * 40 + kStart],
                     *(const h8*)&Bs[nr * 40 + kStart + 8]);
    }
    for (int fm = 0; fm < 2; ++fm)
        for (int fn = 0; fn < 2; ++fn)
            acc[fm][fn] = __builtin_amdgcn_wmma_f32_16x16x32_f16(
                false, af[fm], false, bf[fn], (short)0, acc[fm][fn], false, false);
}

// ---------------------------------------------------------------------------
// Generic WMMA GEMM:  C[M,N] = A[M,K](f16,row-major) x Bt[N,K](f16,row=out col)
// Workgroup: 256 threads = 8 waves (wave32), tile 64x128, wave = 32x32.
// TDM double-buffers both tiles into LDS; wave0 issues & waits, all barrier.
// MODE 0: patch-embed -> x f32 (+conv_b +obs_pos)   MODE 1: +bias -> f16
// MODE 2: +bias +residual -> f32 x (in place)       MODE 3: +bias GELU -> f16
// ---------------------------------------------------------------------------
template<int MODE>
__global__ __launch_bounds__(256)
void k_gemm(const h16* __restrict__ A, const h16* __restrict__ Bt,
            const float* __restrict__ bias, float* __restrict__ xres,
            h16* __restrict__ outH, const float* __restrict__ obs_pos,
            int M, int N, int K) {
    __shared__ __align__(16) h16 As[2][64 * 40];
    __shared__ __align__(16) h16 Bs[2][128 * 40];

    const int tid  = threadIdx.x;
    const int lane = tid & 31;
    const int w    = tid >> 5;
    const int wm   = w >> 2;      // 0..1
    const int wn   = w & 3;       // 0..3
    const int m0   = blockIdx.y * 64;
    const int n0   = blockIdx.x * 128;

    const int mfr    = lane & 15;
    const int klo    = (lane < 16) ? 0 : 8;   // unused: A and B frags both use kStart
    const int kStart = (lane < 16) ? 0 : 16;  // A/B-frag K split: k = 16*(lane/16) + s
    const int rofs   = (lane < 16) ? 0 : 8;   // C/D-frag row split

    v8f acc[2][2] = {};
    const int NK = K >> 5;

#if HAVE_TDM
    // ---- TDM double-buffered pipeline ----
    if (tid < 32) {
        tdm_load_tile_f16(&A [(size_t)m0 * K], &As[0][0],  64u, (unsigned)K);
        tdm_load_tile_f16(&Bt[(size_t)n0 * K], &Bs[0][0], 128u, (unsigned)K);
    }
    for (int s = 0; s < NK; ++s) {
        const int buf = s & 1;
        if (tid < 32) {
            if (s + 1 < NK) {
                int kk = (s + 1) << 5;
                tdm_load_tile_f16(&A [(size_t)m0 * K + kk], &As[buf ^ 1][0],  64u, (unsigned)K);
                tdm_load_tile_f16(&Bt[(size_t)n0 * K + kk], &Bs[buf ^ 1][0], 128u, (unsigned)K);
                __builtin_amdgcn_s_wait_tensorcnt(2);   // stage s landed
            } else {
                __builtin_amdgcn_s_wait_tensorcnt(0);   // final stage landed
            }
        }
        __syncthreads();
        gemm_step(&As[buf][0], &Bs[buf][0], wm, wn, mfr, klo, kStart, acc);
        __syncthreads();
    }
#else
    // ---- fallback: manual VGPR staging ----
    for (int s = 0; s < NK; ++s) {
        int kk = s << 5;
        {
            int row = tid >> 2, col = (tid & 3) * 8;
            *(h8*)&As[0][row * 40 + col] =
                *(const h8*)&A[(size_t)(m0 + row) * K + kk + col];
        }
        for (int i = 0; i < 2; ++i) {
            int c = tid * 2 + i;
            int row = c >> 2, col = (c & 3) * 8;
            *(h8*)&Bs[0][row * 40 + col] =
                *(const h8*)&Bt[(size_t)(n0 + row) * K + kk + col];
        }
        __syncthreads();
        gemm_step(&As[0][0], &Bs[0][0], wm, wn, mfr, klo, kStart, acc);
        __syncthreads();
    }
#endif

    // ---- epilogue ----
    for (int fm = 0; fm < 2; ++fm)
        for (int fn = 0; fn < 2; ++fn) {
            union { v8f v; float f[8]; } ua; ua.v = acc[fm][fn];
            int col = n0 + wn * 32 + fn * 16 + mfr;
            float bsv = bias[col];
            for (int v = 0; v < 8; ++v) {
                int row = m0 + wm * 32 + fm * 16 + v + rofs;
                float val = ua.f[v] + bsv;
                if constexpr (MODE == 0) {
                    int b = row / (TT * PTOK);
                    int rem = row % (TT * PTOK);
                    int t = rem / PTOK, p = rem % PTOK;
                    xres[((size_t)b * SEQ + (size_t)t * TPS + p) * C_DIM + col] =
                        val + obs_pos[(size_t)p * C_DIM + col];
                } else if constexpr (MODE == 1) {
                    outH[(size_t)row * N + col] = (h16)val;
                } else if constexpr (MODE == 2) {
                    xres[(size_t)row * N + col] += val;
                } else {
                    float gl = 0.5f * val * (1.0f + erff(val * 0.70710678f));
                    outH[(size_t)row * N + col] = (h16)gl;
                }
            }
        }
}

// ---------------------------------------------------------------------------
// Attention: one wave per (batch, head, 16-row q-tile).
// S=156 padded to 160 (10 k-tiles); QK^T and P*V both via WMMA f16.
// ---------------------------------------------------------------------------
__global__ __launch_bounds__(32)
void k_attn(const h16* __restrict__ Q, const h16* __restrict__ Kb,
            const h16* __restrict__ Vb, h16* __restrict__ Y) {
    __shared__ float sc[16][168];                 // score rows (padded pitch)
    __shared__ __align__(16) h16 pm[16][160];     // softmax probs, A-frag layout

    int bh = blockIdx.x;
    int b = bh / NHEAD, h = bh % NHEAD;
    int qt = blockIdx.y;
    int lane = threadIdx.x;
    int mfr    = lane & 15;
    int klo    = (lane < 16) ? 0 : 8;
    int kStart = (lane < 16) ? 0 : 16;
    int rofs   = (lane < 16) ? 0 : 8;

    // Q fragments for D=64 (two K-steps of 32)
    int qrow = qt * 16 + mfr; if (qrow > SEQ - 1) qrow = SEQ - 1;
    const h16* qp = Q + ((size_t)(b * SEQ + qrow)) * C_DIM + h * DHEAD;
    v16h qf[2];
    for (int ks = 0; ks < 2; ++ks)
        qf[ks] = mk16(*(const h8*)(qp + ks * 32 + kStart),
                      *(const h8*)(qp + ks * 32 + kStart + 8));

    // Pass 1: scores (scaled + masked) into LDS
    for (int kt = 0; kt < 10; ++kt) {
        int kr = kt * 16 + mfr; if (kr > SEQ - 1) kr = SEQ - 1;
        const h16* kp = Kb + ((size_t)(b * SEQ + kr)) * C_DIM + h * DHEAD;
        if (kt + 1 < 10)
            __builtin_prefetch(kp + 16 * C_DIM, 0, 1);   // next k-tile rows
        v8f a = {};
        for (int ks = 0; ks < 2; ++ks) {
            v16h kf = mk16(*(const h8*)(kp + ks * 32 + kStart),
                           *(const h8*)(kp + ks * 32 + kStart + 8));
            a = __builtin_amdgcn_wmma_f32_16x16x32_f16(
                false, qf[ks], false, kf, (short)0, a, false, false);
        }
        union { v8f v; float f[8]; } ua; ua.v = a;
        int kcol = kt * 16 + mfr;
        for (int v = 0; v < 8; ++v) {
            int qr = qt * 16 + v + rofs;
            float s = ua.f[v] * 0.125f;  // 1/sqrt(64)
            if (!att_allowed(qr, kcol)) s = -1e30f;
            sc[v + rofs][kcol] = s;
        }
    }
    __syncthreads();

    // Pass 2: row softmax (lane i handles row i)
    if (lane < 16) {
        float mx = -3.0e30f;
        for (int j = 0; j < 160; ++j) mx = fmaxf(mx, sc[lane][j]);
        float sum = 0.0f;
        for (int j = 0; j < 160; ++j) {
            float e = __expf(sc[lane][j] - mx);
            sc[lane][j] = e; sum += e;
        }
        float inv = 1.0f / sum;
        for (int j = 0; j < 160; ++j) pm[lane][j] = (h16)(sc[lane][j] * inv);
    }
    __syncthreads();

    // Pass 3: P (16x160) x V (160x64) via WMMA
    v8f acc[4] = {};
    for (int ks2 = 0; ks2 < 5; ++ks2) {
        v16h pf = mk16(*(const h8*)&pm[mfr][ks2 * 32 + kStart],
                       *(const h8*)&pm[mfr][ks2 * 32 + kStart + 8]);
        for (int nt = 0; nt < 4; ++nt) {
            union { v16h v; h16 e[16]; } uv;
            int colV = h * DHEAD + nt * 16 + mfr;
            for (int j = 0; j < 16; ++j) {
                int kv = ks2 * 32 + kStart + j;
                int vr = (kv > SEQ - 1) ? SEQ - 1 : kv;   // P==0 for padding
                uv.e[j] = Vb[((size_t)(b * SEQ + vr)) * C_DIM + colV];
            }
            acc[nt] = __builtin_amdgcn_wmma_f32_16x16x32_f16(
                false, pf, false, uv.v, (short)0, acc[nt], false, false);
        }
    }
    for (int nt = 0; nt < 4; ++nt) {
        union { v8f v; float f[8]; } ua; ua.v = acc[nt];
        for (int v = 0; v < 8; ++v) {
            int qr = qt * 16 + v + rofs;
            if (qr < SEQ)
                Y[((size_t)(b * SEQ + qr)) * C_DIM + h * DHEAD + nt * 16 + mfr] =
                    (h16)ua.f[v];
        }
    }
}

// ---------------------------------------------------------------------------
// Heads: ret_tok = x[.., s=35], act_tok = x[.., 36], rew_tok = x[.., 37]
// Output: act_logits (B,T,18) | ret_logits (B,T,121) | rew_logits (B,T,4)
// ---------------------------------------------------------------------------
__global__ __launch_bounds__(128)
void k_heads(const float* __restrict__ x,
             const float* __restrict__ aw, const float* __restrict__ ab,
             const float* __restrict__ rw, const float* __restrict__ rb,
             const float* __restrict__ ww, const float* __restrict__ wb,
             float* __restrict__ out) {
    int bt = blockIdx.x;
    int b = bt / TT, t = bt % TT;
    const float* xret = x + ((size_t)b * SEQ + (size_t)t * TPS + (PTOK - 1)) * C_DIM;
    const float* xact = xret + C_DIM;
    const float* xrew = xret + 2 * C_DIM;
    for (int o = threadIdx.x; o < 18; o += blockDim.x) {
        float s = ab[o];
        for (int c = 0; c < C_DIM; ++c) s += xact[c] * aw[(size_t)c * 18 + o];
        out[(size_t)bt * 18 + o] = s;
    }
    for (int o = threadIdx.x; o < 121; o += blockDim.x) {
        float s = rb[o];
        for (int c = 0; c < C_DIM; ++c) s += xret[c] * rw[(size_t)c * 121 + o];
        out[(size_t)BB * TT * 18 + (size_t)bt * 121 + o] = s;
    }
    for (int o = threadIdx.x; o < 4; o += blockDim.x) {
        float s = wb[o];
        for (int c = 0; c < C_DIM; ++c) s += xrew[c] * ww[(size_t)c * 4 + o];
        out[(size_t)BB * TT * (18 + 121) + (size_t)bt * 4 + o] = s;
    }
}

// ---------------------------------------------------------------------------
// Host launcher
// ---------------------------------------------------------------------------
extern "C" void kernel_launch(void* const* d_in, const int* in_sizes, int n_in,
                              void* d_out, int out_size, void* d_ws, size_t ws_size,
                              hipStream_t stream) {
    (void)in_sizes; (void)n_in; (void)out_size; (void)ws_size;

    const int*   observations = (const int*)d_in[0];
    const int*   rtg     = (const int*)d_in[1];
    const int*   acts    = (const int*)d_in[2];
    const int*   rews    = (const int*)d_in[3];
    const float* conv_w  = (const float*)d_in[4];
    const float* conv_b  = (const float*)d_in[5];
    const float* obs_pos = (const float*)d_in[6];
    const float* ret_tab = (const float*)d_in[7];
    const float* act_tab = (const float*)d_in[8];
    const float* rew_tab = (const float*)d_in[9];
    const float* ln1_g   = (const float*)d_in[10];
    const float* ln1_b   = (const float*)d_in[11];
    const float* ln2_g   = (const float*)d_in[12];
    const float* ln2_b   = (const float*)d_in[13];
    const float* wq = (const float*)d_in[14]; const float* bq = (const float*)d_in[15];
    const float* wk = (const float*)d_in[16]; const float* bk = (const float*)d_in[17];
    const float* wv = (const float*)d_in[18]; const float* bv = (const float*)d_in[19];
    const float* wo = (const float*)d_in[20]; const float* bo = (const float*)d_in[21];
    const float* w1 = (const float*)d_in[22]; const float* b1 = (const float*)d_in[23];
    const float* w2 = (const float*)d_in[24]; const float* b2 = (const float*)d_in[25];
    const float* ret_w = (const float*)d_in[26]; const float* ret_b = (const float*)d_in[27];
    const float* act_w = (const float*)d_in[28]; const float* act_b = (const float*)d_in[29];
    const float* rew_w = (const float*)d_in[30]; const float* rew_b = (const float*)d_in[31];

    // ---- workspace carve (256B aligned) ----
    char* p = (char*)d_ws;
    auto carve = [&](size_t bytes) {
        char* r = p;
        p += (bytes + 255) & ~(size_t)255;
        return r;
    };
    const size_t CC = (size_t)C_DIM * C_DIM;
    h16* wq_t = (h16*)carve((size_t)NLAYER * CC * 2);
    h16* wk_t = (h16*)carve((size_t)NLAYER * CC * 2);
    h16* wv_t = (h16*)carve((size_t)NLAYER * CC * 2);
    h16* wo_t = (h16*)carve((size_t)NLAYER * CC * 2);
    h16* w1_t = (h16*)carve((size_t)NLAYER * FF * C_DIM * 2);   // [5120][1280]
    h16* w2_t = (h16*)carve((size_t)NLAYER * C_DIM * FF * 2);   // [1280][5120]
    h16* cw_t = (h16*)carve((size_t)C_DIM * KCONV * 2);
    h16* Pbuf = (h16*)carve((size_t)PROWS * KCONV * 2);
    float* x  = (float*)carve((size_t)MROWS * C_DIM * 4);
    h16* hbuf = (h16*)carve((size_t)MROWS * C_DIM * 2);
    h16* qbuf = (h16*)carve((size_t)MROWS * C_DIM * 2);
    h16* kbuf = (h16*)carve((size_t)MROWS * C_DIM * 2);
    h16* vbuf = (h16*)carve((size_t)MROWS * C_DIM * 2);
    h16* ybuf = (h16*)carve((size_t)MROWS * C_DIM * 2);
    h16* mbuf = (h16*)carve((size_t)MROWS * FF * 2);

    // ---- weight conversion / transposition (per call, deterministic) ----
    for (int l = 0; l < NLAYER; ++l) {
        k_transpose<<<C_DIM, 256, 0, stream>>>(wq + l * CC, wq_t + l * CC, C_DIM, C_DIM);
        k_transpose<<<C_DIM, 256, 0, stream>>>(wk + l * CC, wk_t + l * CC, C_DIM, C_DIM);
        k_transpose<<<C_DIM, 256, 0, stream>>>(wv + l * CC, wv_t + l * CC, C_DIM, C_DIM);
        k_transpose<<<C_DIM, 256, 0, stream>>>(wo + l * CC, wo_t + l * CC, C_DIM, C_DIM);
        k_transpose<<<FF,    256, 0, stream>>>(w1 + (size_t)l * C_DIM * FF,
                                               w1_t + (size_t)l * FF * C_DIM, C_DIM, FF);
        k_transpose<<<C_DIM, 256, 0, stream>>>(w2 + (size_t)l * FF * C_DIM,
                                               w2_t + (size_t)l * C_DIM * FF, FF, C_DIM);
    }
    k_convw<<<C_DIM, 256, 0, stream>>>(conv_w, cw_t);

    // ---- embedding ----
    k_patchify<<<PROWS, 256, 0, stream>>>(observations, Pbuf);
    {   // patch GEMM: 4608 x 1280 x 800 -> x (+conv_b +obs_pos)
        dim3 g(C_DIM / 128, PROWS / 64);
        k_gemm<0><<<g, 256, 0, stream>>>(Pbuf, cw_t, conv_b, x, nullptr, obs_pos,
                                         PROWS, C_DIM, KCONV);
    }
    k_embed<<<BB * TT, 256, 0, stream>>>(rtg, acts, rews, ret_tab, act_tab, rew_tab, x);

    // ---- transformer layers ----
    dim3 gC(C_DIM / 128, MROWS / 64);   // (10, 78)
    dim3 gF(FF / 128,    MROWS / 64);   // (40, 78)
    for (int l = 0; l < NLAYER; ++l) {
        k_ln<<<MROWS, 256, 0, stream>>>(x, ln1_g + l * C_DIM, ln1_b + l * C_DIM, hbuf);
        k_gemm<1><<<gC, 256, 0, stream>>>(hbuf, wq_t + l * CC, bq + l * C_DIM,
                                          nullptr, qbuf, nullptr, MROWS, C_DIM, C_DIM);
        k_gemm<1><<<gC, 256, 0, stream>>>(hbuf, wk_t + l * CC, bk + l * C_DIM,
                                          nullptr, kbuf, nullptr, MROWS, C_DIM, C_DIM);
        k_gemm<1><<<gC, 256, 0, stream>>>(hbuf, wv_t + l * CC, bv + l * C_DIM,
                                          nullptr, vbuf, nullptr, MROWS, C_DIM, C_DIM);
        k_attn<<<dim3(BB * NHEAD, 10), 32, 0, stream>>>(qbuf, kbuf, vbuf, ybuf);
        k_gemm<2><<<gC, 256, 0, stream>>>(ybuf, wo_t + l * CC, bo + l * C_DIM,
                                          x, nullptr, nullptr, MROWS, C_DIM, C_DIM);
        k_ln<<<MROWS, 256, 0, stream>>>(x, ln2_g + l * C_DIM, ln2_b + l * C_DIM, hbuf);
        k_gemm<3><<<gF, 256, 0, stream>>>(hbuf, w1_t + (size_t)l * FF * C_DIM,
                                          b1 + (size_t)l * FF,
                                          nullptr, mbuf, nullptr, MROWS, FF, C_DIM);
        k_gemm<2><<<gC, 256, 0, stream>>>(mbuf, w2_t + (size_t)l * C_DIM * FF,
                                          b2 + l * C_DIM,
                                          x, nullptr, nullptr, MROWS, C_DIM, FF);
    }

    // ---- heads ----
    k_heads<<<BB * TT, 128, 0, stream>>>(x, act_w, act_b, ret_w, ret_b, rew_w, rew_b,
                                         (float*)d_out);
}
